// AttentionGCNLayer_38448547233854
// MI455X (gfx1250) — matrix operand order repaired and checked by reading, not compiled, not run on hardware
//
#include <hip/hip_runtime.h>
#include <hip/hip_bf16.h>

// ---------------- problem constants ----------------
#define BB    8
#define NN    1024
#define CIN   128
#define COUT  256
#define NHEAD 8
#define HD    32
#define MROWS (BB * NN)   // 8192

typedef __attribute__((ext_vector_type(16))) __bf16 v16bf;
typedef __attribute__((ext_vector_type(8)))  __bf16 v8bf;
typedef __attribute__((ext_vector_type(8)))  float  v8f;

// ---------------- WMMA helpers ----------------
__device__ inline v8f wmma_bf16(v16bf a, v16bf b, v8f c) {
    return __builtin_amdgcn_wmma_f32_16x16x32_bf16(
        false, a, false, b, (short)0, c, false, false);
}

// Two contiguous 16-byte loads -> one 16-element fragment.
__device__ inline v16bf load2x8(const __bf16* p0, const __bf16* p1) {
    v8bf lo = *(const v8bf*)p0;
    v8bf hi = *(const v8bf*)p1;
    v16bf r;
#pragma unroll
    for (int i = 0; i < 8; ++i) { r[i] = lo[i]; r[i + 8] = hi[i]; }
    return r;
}

// A fragment: 16x32 bf16 row-major tile.
// lane 0-15 : M=lane,    elems 0-7 -> K=0..7,  elems 8-15 -> K=16..23
// lane 16-31: M=lane-16, elems 0-7 -> K=8..15, elems 8-15 -> K=24..31
__device__ inline v16bf load_a_frag(const __bf16* A, int lda, int lane) {
    const __bf16* row = A + (long)(lane & 15) * lda + ((lane & 16) ? 8 : 0);
    return load2x8(row, row + 16);
}

// B fragment from TRANSPOSED operand Bt[N][K]; ptr pre-offset with n*ldt + kk + kb.
__device__ inline v16bf load_bt_frag(const __bf16* p) {
    return load2x8(p, p + 16);
}

// wave32 butterfly sum
__device__ inline float wave_sum(float v) {
#pragma unroll
    for (int o = 16; o > 0; o >>= 1) v += __shfl_xor(v, o, 32);
    return v;
}

// ---------------- conversion / packing kernels ----------------
__global__ void cvt_f32_bf16(const float* __restrict__ in, __bf16* __restrict__ out, long n) {
    long i = (long)blockIdx.x * blockDim.x + threadIdx.x;
    long stride = (long)gridDim.x * blockDim.x;
    for (; i < n; i += stride) out[i] = (__bf16)in[i];
}

// f32 [rows,cols] -> bf16 TRANSPOSED: out[(co+c)*ldo + r]
__global__ void cvt2d_f32_bf16_t(const float* __restrict__ in, __bf16* __restrict__ out,
                                 int rows, int cols, int ldo, int co) {
    int i = blockIdx.x * blockDim.x + threadIdx.x;
    if (i < rows * cols) {
        int r = i / cols, c = i - r * cols;
        out[(long)(co + c) * ldo + r] = (__bf16)in[i];
    }
}

__global__ void copy_f32(const float* __restrict__ in, float* __restrict__ out, int n) {
    int i = blockIdx.x * blockDim.x + threadIdx.x;
    if (i < n) out[i] = in[i];
}

// bf16 batched transpose: in [nb][R][C] -> out [nb][C][R]  (coalesced writes)
__global__ void transpose_bf16(const __bf16* __restrict__ in, __bf16* __restrict__ out,
                               int nb, int R, int C) {
    long i = (long)blockIdx.x * blockDim.x + threadIdx.x;
    long per = (long)R * C;
    if (i < nb * per) {
        long b = i / per;
        long rem = i - b * per;
        int c = (int)(rem / R);
        int r = (int)(rem - (long)c * R);
        out[i] = in[b * per + (long)r * C + c];
    }
}

// vT[(b*H+h)*HD + d][n] = qkv[(b*NN+n)*768 + 512 + h*HD + d]
__global__ void build_vT(const __bf16* __restrict__ qkv, __bf16* __restrict__ vT) {
    long i = (long)blockIdx.x * blockDim.x + threadIdx.x;
    if (i < (long)BB * NHEAD * HD * NN) {
        int  n  = (int)(i & (NN - 1));
        long t  = i >> 10;            // (b*H+h)*HD + d
        int  d  = (int)(t & (HD - 1));
        long bh = t >> 5;             // b*H + h
        int  h  = (int)(bh & (NHEAD - 1));
        long b  = bh >> 3;
        vT[i] = qkv[(b * NN + n) * 768 + 2 * COUT + h * HD + d];
    }
}

// ---------------- WMMA GEMM: each wave -> 16x64 output strip ----------------
// Software-pipelined: fragments for step k+1 are loaded before step k's WMMAs.
__global__ __launch_bounds__(128) void gemm_bf16_wmma(
    const __bf16* __restrict__ A,  long aBatch, int lda,
    const __bf16* __restrict__ Bt, long bBatch, int ldt,
    const float*  __restrict__ bias,
    const float*  __restrict__ residual,
    float* __restrict__ outF, __bf16* __restrict__ outBf,
    int M, int Nn, int K, int rowsPerBatch, int applyRelu)
{
    int lane = threadIdx.x & 31;
    int wave = threadIdx.x >> 5;
    int nstrips = Nn >> 6;                      // 64-wide strips
    int strip = blockIdx.x * 4 + wave;
    int total = (M >> 4) * nstrips;
    if (strip >= total) return;

    int tm = strip / nstrips;
    int ts = strip - tm * nstrips;
    int gm0 = tm << 4;
    int batch = gm0 / rowsPerBatch;
    int rib   = gm0 - batch * rowsPerBatch;

    int akb = (lane & 16) ? 8 : 0;
    int bkb = (lane & 16) ? 8 : 0;
    const __bf16* Arow = A + (long)batch * aBatch + (long)(rib + (lane & 15)) * lda + akb;
    const __bf16* Btb  = Bt + (long)batch * bBatch
                       + (long)(ts * 64 + (lane & 15)) * ldt + bkb;

    v8f acc0 = {}, acc1 = {}, acc2 = {}, acc3 = {};

    // prologue: load step-0 fragments
    v16bf af = load2x8(Arow, Arow + 16);
    v16bf b0 = load_bt_frag(Btb);
    v16bf b1 = load_bt_frag(Btb + (long)16 * ldt);
    v16bf b2 = load_bt_frag(Btb + (long)32 * ldt);
    v16bf b3 = load_bt_frag(Btb + (long)48 * ldt);

    for (int kk = 32; kk < K; kk += 32) {
        __builtin_prefetch(Arow + kk + 224, 0, 3);     // WGP-scope prefetch
        v16bf afn = load2x8(Arow + kk, Arow + kk + 16);
        v16bf b0n = load_bt_frag(Btb + kk);
        v16bf b1n = load_bt_frag(Btb + (long)16 * ldt + kk);
        v16bf b2n = load_bt_frag(Btb + (long)32 * ldt + kk);
        v16bf b3n = load_bt_frag(Btb + (long)48 * ldt + kk);

        acc0 = wmma_bf16(af, b0, acc0);
        acc1 = wmma_bf16(af, b1, acc1);
        acc2 = wmma_bf16(af, b2, acc2);
        acc3 = wmma_bf16(af, b3, acc3);

        af = afn; b0 = b0n; b1 = b1n; b2 = b2n; b3 = b3n;
    }
    acc0 = wmma_bf16(af, b0, acc0);
    acc1 = wmma_bf16(af, b1, acc1);
    acc2 = wmma_bf16(af, b2, acc2);
    acc3 = wmma_bf16(af, b3, acc3);

    int mb = (lane & 16) ? 8 : 0;
    int gn0 = (ts << 6) + (lane & 15);
    v8f accs[4] = {acc0, acc1, acc2, acc3};
#pragma unroll
    for (int j = 0; j < 4; ++j) {
        int gn = gn0 + j * 16;
        float bv = bias ? bias[gn] : 0.0f;
#pragma unroll
        for (int r = 0; r < 8; ++r) {
            long idx = (long)(gm0 + mb + r) * Nn + gn;
            float v = accs[j][r] + bv;
            if (applyRelu) v = fmaxf(v, 0.0f);
            if (residual)  v += residual[idx];
            if (outF)  outF[idx]  = v;
            if (outBf) outBf[idx] = (__bf16)v;
        }
    }
}

// ---------------- LayerNorm over C=256 channels (shuffle reductions) ----------------
__global__ __launch_bounds__(COUT) void ln_kernel(
    const float* __restrict__ in, const float* __restrict__ gamma,
    const float* __restrict__ beta,
    float* __restrict__ outF, __bf16* __restrict__ outBf)
{
    __shared__ float wsum[8];
    int row = blockIdx.x;
    int c   = threadIdx.x;
    int wid = c >> 5;
    int lane = c & 31;
    long idx = (long)row * COUT + c;
    float v = in[idx];

    float s = wave_sum(v);
    if (lane == 0) wsum[wid] = s;
    __syncthreads();
    float mu = 0.0f;
#pragma unroll
    for (int i = 0; i < 8; ++i) mu += wsum[i];
    mu *= (1.0f / COUT);
    __syncthreads();

    float d = v - mu;
    float s2 = wave_sum(d * d);
    if (lane == 0) wsum[wid] = s2;
    __syncthreads();
    float var = 0.0f;
#pragma unroll
    for (int i = 0; i < 8; ++i) var += wsum[i];
    var *= (1.0f / COUT);

    float y = d * rsqrtf(var + 1e-5f) * gamma[c] + beta[c];
    if (outF)  outF[idx]  = y;
    if (outBf) outBf[idx] = (__bf16)y;
}

// ---------------- attention: one wave per (b, head, 16-query tile) ----------------
// qkv: [B*N,768] bf16 (q@0, k@256).  vT: [(b*H+h)*HD + d][NN] bf16.
__global__ __launch_bounds__(32) void attn_kernel(
    const __bf16* __restrict__ qkv, const __bf16* __restrict__ vT,
    __bf16* __restrict__ attout)
{
    __shared__ float sc[16 * NN];               // exactly 64 KB
    int lane = threadIdx.x;
    int blk  = blockIdx.x;                      // b*(H*64) + h*64 + mt
    int mt = blk & 63;
    int h  = (blk >> 6) & (NHEAD - 1);
    int b  = blk >> 9;
    long rowBase = (long)b * NN;
    long bh = (long)b * NHEAD + h;

    // Q tile 16x32 -> one A fragment (M=16 queries, K=D=32)
    const __bf16* qbase = qkv + (rowBase + mt * 16) * 768 + h * HD;
    v16bf qa = load_a_frag(qbase, 768, lane);

    int nsel = lane & 15;
    int mb   = (lane & 16) ? 8 : 0;
    long kstride = 768;
    const __bf16* kcol = qkv + (rowBase + nsel) * kstride
                       + COUT + h * HD + ((lane & 16) ? 8 : 0);

    // scores = Q @ K^T : 64 WMMAs, K-fragment for nt+1 preloaded before WMMA nt
    v16bf kb = load2x8(kcol, kcol + 16);
    for (int nt = 0; nt < 64; ++nt) {
        v16bf kbn = kb;
        if (nt < 63) {
            const __bf16* kp = kcol + (long)(nt + 1) * 16 * kstride;
            kbn = load2x8(kp, kp + 16);
        }
        v8f c = {};
        c = wmma_bf16(qa, kb, c);
        int dn = nt * 16 + nsel;
#pragma unroll
        for (int r = 0; r < 8; ++r) sc[(mb + r) * NN + dn] = c[r];
        kb = kbn;
    }
    __syncthreads();

    // parallel softmax: lane handles row (lane&15), half (lane>>4)
    const float scale = 0.17677669529663687f;   // 1/sqrt(32)
    float pinv;
    {
        float* row = sc + (lane & 15) * NN;
        int j0 = (lane >> 4) * (NN / 2);
        float mx = -1e30f;
        for (int j = 0; j < NN / 2; ++j) mx = fmaxf(mx, row[j0 + j]);
        mx = fmaxf(mx, __shfl_xor(mx, 16, 32));
        float s = 0.0f;
        for (int j = 0; j < NN / 2; ++j) {
            float e = __expf((row[j0 + j] - mx) * scale);
            row[j0 + j] = e; s += e;
        }
        s += __shfl_xor(s, 16, 32);
        pinv = 1.0f / s;                         // row mapping == A-frag row mapping
    }
    __syncthreads();

    // out = P(16x1024, f32 in LDS, cvt on the fly) @ V(1024x32 via vT), pipelined
    const float* prow = sc + (lane & 15) * NN + ((lane & 16) ? 8 : 0);
    int vkb = (lane & 16) ? 8 : 0;
    const __bf16* v0base = vT + (bh * HD + nsel) * NN + vkb;
    const __bf16* v1base = v0base + (long)16 * NN;

    auto make_pa = [&](int kk) {
        const float* p0 = prow + kk;
        const float* p1 = prow + kk + 16;
        v16bf pa;
#pragma unroll
        for (int e = 0; e < 8; ++e) {
            pa[e]     = (__bf16)(p0[e] * pinv);
            pa[e + 8] = (__bf16)(p1[e] * pinv);
        }
        return pa;
    };

    v8f acc0 = {}, acc1 = {};
    v16bf pa  = make_pa(0);
    v16bf vb0 = load2x8(v0base, v0base + 16);
    v16bf vb1 = load2x8(v1base, v1base + 16);
    for (int kk = 32; kk < NN; kk += 32) {
        v16bf pan  = make_pa(kk);
        v16bf vb0n = load2x8(v0base + kk, v0base + kk + 16);
        v16bf vb1n = load2x8(v1base + kk, v1base + kk + 16);
        acc0 = wmma_bf16(pa, vb0, acc0);
        acc1 = wmma_bf16(pa, vb1, acc1);
        pa = pan; vb0 = vb0n; vb1 = vb1n;
    }
    acc0 = wmma_bf16(pa, vb0, acc0);
    acc1 = wmma_bf16(pa, vb1, acc1);

#pragma unroll
    for (int r = 0; r < 8; ++r) {
        long ridx = (rowBase + mt * 16 + mb + r) * COUT + h * HD;
        attout[ridx + nsel]      = (__bf16)acc0[r];
        attout[ridx + 16 + nsel] = (__bf16)acc1[r];
    }
}

// ---------------- host-side launch ----------------
extern "C" void kernel_launch(void* const* d_in, const int* in_sizes, int n_in,
                              void* d_out, int out_size, void* d_ws, size_t ws_size,
                              hipStream_t stream) {
    const float* x    = (const float*)d_in[0];
    const float* adj  = (const float*)d_in[1];
    const float* W_it = (const float*)d_in[2];
    const float* b_it = (const float*)d_in[3];
    const float* W_g  = (const float*)d_in[4];
    const float* b_g  = (const float*)d_in[5];
    const float* W_q  = (const float*)d_in[6];
    const float* b_q  = (const float*)d_in[7];
    const float* W_k  = (const float*)d_in[8];
    const float* b_k  = (const float*)d_in[9];
    const float* W_v  = (const float*)d_in[10];
    const float* b_v  = (const float*)d_in[11];
    const float* W_o  = (const float*)d_in[12];
    const float* b_o  = (const float*)d_in[13];
    const float* g1   = (const float*)d_in[14];
    const float* be1  = (const float*)d_in[15];
    const float* g2   = (const float*)d_in[16];
    const float* be2  = (const float*)d_in[17];
    float* out = (float*)d_out;

    char* ws = (char*)d_ws;
    size_t off = 0;
    auto take = [&](size_t bytes) -> char* {
        char* p = ws + off;
        off = (off + bytes + 255) & ~(size_t)255;
        return p;
    };
    __bf16* xbf     = (__bf16*)take((size_t)MROWS * CIN * 2);
    __bf16* adjbf   = (__bf16*)take((size_t)BB * NN * NN * 2);
    __bf16* witT    = (__bf16*)take((size_t)CIN * COUT * 2);       // [256][128]
    __bf16* wgT     = (__bf16*)take((size_t)CIN * COUT * 2);       // [256][128]
    __bf16* wqkvT   = (__bf16*)take((size_t)COUT * 3 * COUT * 2);  // [768][256]
    __bf16* woT     = (__bf16*)take((size_t)COUT * COUT * 2);      // [256][256]
    float*  bqkvf   = (float*) take((size_t)3 * COUT * 4);
    float*  identf  = (float*) take((size_t)MROWS * COUT * 4);
    __bf16* glinbf  = (__bf16*)take((size_t)MROWS * COUT * 2);
    __bf16* glinT   = (__bf16*)take((size_t)MROWS * COUT * 2);     // [b][256][1024]
    float*  hf      = (float*) take((size_t)MROWS * COUT * 4);
    __bf16* hbf     = (__bf16*)take((size_t)MROWS * COUT * 2);
    __bf16* qkvbf   = (__bf16*)take((size_t)MROWS * 3 * COUT * 2);
    __bf16* vTbf    = (__bf16*)take((size_t)BB * NHEAD * HD * NN * 2);
    __bf16* attbf   = (__bf16*)take((size_t)MROWS * COUT * 2);
    (void)ws_size; (void)in_sizes; (void)n_in; (void)out_size;

    // --- stage 0: conversions (weights transposed for contiguous B-fragments) ---
    cvt_f32_bf16<<<4096, 256, 0, stream>>>(x,   xbf,   (long)MROWS * CIN);
    cvt_f32_bf16<<<8192, 256, 0, stream>>>(adj, adjbf, (long)BB * NN * NN);
    cvt2d_f32_bf16_t<<<(CIN*COUT + 255)/256, 256, 0, stream>>>(W_it, witT, CIN, COUT, CIN, 0);
    cvt2d_f32_bf16_t<<<(CIN*COUT + 255)/256, 256, 0, stream>>>(W_g,  wgT,  CIN, COUT, CIN, 0);
    cvt2d_f32_bf16_t<<<(COUT*COUT + 255)/256, 256, 0, stream>>>(W_q, wqkvT, COUT, COUT, COUT, 0);
    cvt2d_f32_bf16_t<<<(COUT*COUT + 255)/256, 256, 0, stream>>>(W_k, wqkvT, COUT, COUT, COUT, COUT);
    cvt2d_f32_bf16_t<<<(COUT*COUT + 255)/256, 256, 0, stream>>>(W_v, wqkvT, COUT, COUT, COUT, 2*COUT);
    cvt2d_f32_bf16_t<<<(COUT*COUT + 255)/256, 256, 0, stream>>>(W_o, woT, COUT, COUT, COUT, 0);
    copy_f32<<<1, 256, 0, stream>>>(b_q, bqkvf,          COUT);
    copy_f32<<<1, 256, 0, stream>>>(b_k, bqkvf + COUT,   COUT);
    copy_f32<<<1, 256, 0, stream>>>(b_v, bqkvf + 2*COUT, COUT);

    // --- stage 1: identity = x@W_it + b_it (f32) ; glin = x@W_g + b_g (bf16) ---
    {
        int blocks = ((MROWS/16) * (COUT/64) + 3) / 4;   // 512
        gemm_bf16_wmma<<<blocks, 128, 0, stream>>>(
            xbf, 0, CIN, witT, 0, CIN, b_it, nullptr,
            identf, nullptr, MROWS, COUT, CIN, MROWS, 0);
        gemm_bf16_wmma<<<blocks, 128, 0, stream>>>(
            xbf, 0, CIN, wgT, 0, CIN, b_g, nullptr,
            nullptr, glinbf, MROWS, COUT, CIN, MROWS, 0);
    }
    transpose_bf16<<<((long)MROWS*COUT + 255)/256, 256, 0, stream>>>(glinbf, glinT, BB, NN, COUT);

    // --- stage 2: ident += relu(adj @ glin)  (batched, K=1024) ---
    {
        int blocks = ((MROWS/16) * (COUT/64) + 3) / 4;
        gemm_bf16_wmma<<<blocks, 128, 0, stream>>>(
            adjbf, (long)NN * NN, NN,
            glinT, (long)COUT * NN, NN,
            nullptr, identf, identf, nullptr,
            MROWS, COUT, NN, NN, 1);
    }

    // --- stage 3: h = LN1(ident) ---
    ln_kernel<<<MROWS, COUT, 0, stream>>>(identf, g1, be1, hf, hbf);

    // --- stage 4: qkv = h @ [Wq|Wk|Wv] + bias  (N=768) ---
    {
        int blocks = ((MROWS/16) * (3*COUT/64) + 3) / 4;  // 1536
        gemm_bf16_wmma<<<blocks, 128, 0, stream>>>(
            hbf, 0, COUT, wqkvT, 0, COUT, bqkvf, nullptr,
            nullptr, qkvbf, MROWS, 3*COUT, COUT, MROWS, 0);
    }
    build_vT<<<((long)BB*NHEAD*HD*NN + 255)/256, 256, 0, stream>>>(qkvbf, vTbf);

    // --- stage 5: multi-head attention ---
    attn_kernel<<<BB * NHEAD * (NN/16), 32, 0, stream>>>(qkvbf, vTbf, attbf);

    // --- stage 6: pre = h + attout@W_o + b_o ---
    {
        int blocks = ((MROWS/16) * (COUT/64) + 3) / 4;
        gemm_bf16_wmma<<<blocks, 128, 0, stream>>>(
            attbf, 0, COUT, woT, 0, COUT, b_o, hf,
            identf, nullptr, MROWS, COUT, COUT, MROWS, 0);
    }

    // --- stage 7: out = LN2(pre) ---
    ln_kernel<<<MROWS, COUT, 0, stream>>>(identf, g2, be2, out, nullptr);
}
